// MultiHeadAttention_12429635355010
// MI455X (gfx1250) — compile-verified, matrix operand order checked by reading, not run on hardware
//
#include <hip/hip_runtime.h>
#include <hip/hip_bf16.h>

typedef _Float16 f16_t;
typedef __attribute__((ext_vector_type(16))) _Float16 v16h;
typedef __attribute__((ext_vector_type(8)))  _Float16 v8h;
typedef __attribute__((ext_vector_type(8)))  float    v8f;

#define D_MODEL 1024
#define N_HEADS 16
#define D_HEAD  64
#define BATCH   4
#define SEQ     2048
#define M_TOTAL (BATCH * SEQ)   /* 8192 rows */

// ---------------------------------------------------------------------------
// Fragment helpers (CDNA5 WMMA 16x16x32 f16, wave32).
// Lane decomposition: r = lane & 15, hf = lane >> 4.
// A/B fragment for lane (hf,r): element e (0..15): g = e/8, j = e%8,
//   contraction index K = g*16 + hf*8 + j, row (M or N) = r.
// => per lane: two contiguous 8 x f16 (16B) chunks at k0+hf*8 and k0+16+hf*8.
// C/D fragment: element i -> row M = hf*8 + i, col N = r.
// ---------------------------------------------------------------------------
static __device__ __forceinline__ v16h make_frag(const f16_t* p) {
  v8h lo = *(const v8h*)(p);
  v8h hi = *(const v8h*)(p + 16);
  return __builtin_shufflevector(lo, hi, 0, 1, 2, 3, 4, 5, 6, 7,
                                         8, 9, 10, 11, 12, 13, 14, 15);
}

static __device__ __forceinline__ v16h load_frag(const f16_t* base, int ld,
                                                 int row, int k0, int hf) {
  return make_frag(base + (size_t)row * ld + k0 + hf * 8);
}

static __device__ __forceinline__ v8f wmma_f16(v16h a, v16h b, v8f c) {
  return __builtin_amdgcn_wmma_f32_16x16x32_f16(false, a, false, b,
                                                (short)0, c, false, false);
}

// ---------------------------------------------------------------------------
// CDNA5 async global->LDS copy (ASYNCcnt tracked).  VDST carries the wave-
// relative LDS byte offset; generic shared pointers put that offset in the low
// 32 address bits (flat aperture scheme), so a truncating cast recovers it.
// ---------------------------------------------------------------------------
static __device__ __forceinline__ void async_copy_b128(unsigned lds_off,
                                                       const void* gaddr) {
  unsigned long long ga = (unsigned long long)gaddr;
  asm volatile("global_load_async_to_lds_b128 %0, %1, off"
               :: "v"(lds_off), "v"(ga)
               : "memory");
}
static __device__ __forceinline__ void async_wait_all() {
  asm volatile("s_wait_asynccnt 0x0" ::: "memory");
}
static __device__ __forceinline__ void async_wait_one_block() {
  // <=32 outstanding: the previously staged 8KB+8KB block (32 b128 ops) has
  // completed while the freshly issued block may still be in flight.
  asm volatile("s_wait_asynccnt 0x20" ::: "memory");
}

// ---------------------------------------------------------------------------
// Precision conversion kernels (fp32 -> f16, optional transpose for weights).
// ---------------------------------------------------------------------------
__global__ __launch_bounds__(256) void cvt_f32_to_f16(
    const float* __restrict__ in, f16_t* __restrict__ out, int n) {
  int i = blockIdx.x * blockDim.x + threadIdx.x;
  if (i < n) out[i] = (f16_t)in[i];
}

// out[n * rows + k] = in[k * cols + n]  (weights stored [K, N] -> [N, K])
__global__ __launch_bounds__(256) void cvt_transpose_f32_to_f16(
    const float* __restrict__ in, f16_t* __restrict__ out, int rows, int cols) {
  int i = blockIdx.x * blockDim.x + threadIdx.x;
  if (i < rows * cols) {
    int k = i / cols;
    int n = i % cols;
    out[(size_t)n * rows + k] = (f16_t)in[i];
  }
}

// ---------------------------------------------------------------------------
// QKV projection: [8192,1024] x [1024,3072] -> Q,K [bh][t][d] and V^T [bh][d][t]
// One wave per block computes a 64x64 output tile: 4x4 WMMA accumulators.
// K-loop unrolled 2x with ping-pong fragment register sets: set X is consumed
// in place while set Y's loads for the next 32-chunk are in flight -- full
// load/WMMA overlap with zero register-rotation moves.
// ---------------------------------------------------------------------------
__global__ __launch_bounds__(32) void qkv_gemm_kernel(
    const f16_t* __restrict__ xh, const f16_t* __restrict__ wt,
    const float* __restrict__ bias,
    f16_t* __restrict__ qb, f16_t* __restrict__ kb, f16_t* __restrict__ vtb) {
  const int lane = threadIdx.x & 31;
  const int r = lane & 15;
  const int hf = lane >> 4;
  const int m0 = blockIdx.x * 64;
  const int n0 = blockIdx.y * 64;

  v8f zero = {};
  v8f acc[4][4];
#pragma unroll
  for (int i = 0; i < 4; ++i)
#pragma unroll
    for (int j = 0; j < 4; ++j) acc[i][j] = zero;

  v16h a0[4], b0[4], a1[4], b1[4];

  auto load_set = [&](v16h* A, v16h* B, int kk) {
#pragma unroll
    for (int i = 0; i < 4; ++i)
      A[i] = load_frag(xh, D_MODEL, m0 + i * 16 + r, kk, hf);
#pragma unroll
    for (int j = 0; j < 4; ++j)
      B[j] = load_frag(wt, D_MODEL, n0 + j * 16 + r, kk, hf);
  };
  auto mma_set = [&](const v16h* A, const v16h* B) {
#pragma unroll
    for (int i = 0; i < 4; ++i)
#pragma unroll
      for (int j = 0; j < 4; ++j)
        acc[i][j] = wmma_f16(A[i], B[j], acc[i][j]);
  };

  load_set(a0, b0, 0);
  for (int k0 = 0; k0 < D_MODEL; k0 += 64) {
    // phase 0: consume set 0, stream set 1 (k0+32 always < D_MODEL here)
    load_set(a1, b1, k0 + 32);
    __builtin_prefetch(xh + (size_t)(m0 + r) * D_MODEL + k0 + 64, 0, 1);
    __builtin_prefetch(wt + (size_t)(n0 + r) * D_MODEL + k0 + 64, 0, 1);
    mma_set(a0, b0);
    // phase 1: consume set 1, stream set 0 for the next outer step
    if (k0 + 64 < D_MODEL) load_set(a0, b0, k0 + 64);
    mma_set(a1, b1);
  }

  // Epilogue: add bias, scatter into per-head Q, K, V^T layouts (f16).
#pragma unroll
  for (int j = 0; j < 4; ++j) {
    const int n = n0 + j * 16 + r;
    const float bv = bias[n];
    const int region = n >> 10;      // 0 = Q, 1 = K, 2 = V
    const int nc = n & 1023;
    const int h = nc >> 6;
    const int d = nc & 63;
#pragma unroll
    for (int i = 0; i < 4; ++i) {
#pragma unroll
      for (int e = 0; e < 8; ++e) {
        const int m = m0 + i * 16 + hf * 8 + e;
        const int bb = m >> 11;      // batch
        const int t = m & 2047;      // seq position
        const f16_t val = (f16_t)(acc[i][j][e] + bv);
        const size_t head = (size_t)(bb * N_HEADS + h);
        if (region == 0)
          qb[(head * SEQ + t) * D_HEAD + d] = val;
        else if (region == 1)
          kb[(head * SEQ + t) * D_HEAD + d] = val;
        else
          vtb[(head * D_HEAD + d) * SEQ + t] = val;
      }
    }
  }
}

// ---------------------------------------------------------------------------
// Causal flash attention: one wave per 16 query rows of one (batch, head).
// Key/value blocks of 64 are staged into double-buffered LDS tiles with
// CDNA5 async global->LDS copies (ASYNCcnt), so block i+1 streams in while
// WMMAs consume block i.  S = QK^T via WMMA, online softmax with wave32 shfl
// reductions, P re-shaped C->A layout through LDS, O += P@V via WMMA vs V^T.
// ---------------------------------------------------------------------------
__global__ __launch_bounds__(32) void attn_kernel(
    const f16_t* __restrict__ qb, const f16_t* __restrict__ kb,
    const f16_t* __restrict__ vtb, f16_t* __restrict__ attn) {
  __shared__ __align__(16) f16_t Kbuf[2][64 * 64];  // 2 x 8KB, rows of 128B
  __shared__ __align__(16) f16_t Vbuf[2][64 * 64];  // 2 x 8KB, rows of 128B
  __shared__ __align__(16) f16_t Pbuf[16][80];      // 64 keys + pad (16B rows)

  const int lane = threadIdx.x & 31;
  const int r = lane & 15;
  const int hf = lane >> 4;
  const int nq = SEQ / 16;                       // 128 q-tiles per (b,h)
  const int bh = blockIdx.x / nq;
  const int q0 = (blockIdx.x % nq) * 16;
  const int b = bh / N_HEADS;
  const int h = bh % N_HEADS;

  const f16_t* Q = qb + (size_t)bh * SEQ * D_HEAD;
  const f16_t* K = kb + (size_t)bh * SEQ * D_HEAD;
  const f16_t* Vt = vtb + (size_t)bh * D_HEAD * SEQ;

  // Hoisted loop-invariant Q fragments (rows q0..q0+15, Dh in two 32-chunks).
  v16h qf[2];
  qf[0] = load_frag(Q, D_HEAD, q0 + r, 0, hf);
  qf[1] = load_frag(Q, D_HEAD, q0 + r, 32, hf);

  // Stage one 64-key block (K tile + V^T tile) into LDS buffer `buf`.
  // K rows are contiguous (ld == D_HEAD) -> flat 8KB copy; V^T rows are
  // 128B slices of stride SEQ.
  auto stage = [&](int buf, int kt0) {
    const char* gK = (const char*)(K + (size_t)kt0 * D_HEAD);
    const unsigned kB = (unsigned)(unsigned long long)&Kbuf[buf][0];
    const unsigned vB = (unsigned)(unsigned long long)&Vbuf[buf][0];
#pragma unroll
    for (int t2 = 0; t2 < 16; ++t2) {
      const unsigned idx = t2 * 32 + lane;       // 16B chunk id (0..511)
      async_copy_b128(kB + idx * 16, gK + idx * 16);
    }
#pragma unroll
    for (int t2 = 0; t2 < 16; ++t2) {
      const unsigned idx = t2 * 32 + lane;
      const unsigned row = idx >> 3;             // d row (0..63)
      const unsigned ch = idx & 7;               // 16B chunk within row
      async_copy_b128(vB + row * 128 + ch * 16,
                      (const char*)(Vt + (size_t)row * SEQ + kt0) + ch * 16);
    }
  };

  float mrow[8], lrow[8];
  v8f zero = {};
  v8f O[4];
#pragma unroll
  for (int i = 0; i < 8; ++i) { mrow[i] = -3.0e38f; lrow[i] = 0.0f; }
#pragma unroll
  for (int jd = 0; jd < 4; ++jd) O[jd] = zero;

  const float scale = 0.125f;                    // 1/sqrt(D_HEAD)
  const int kt_end = q0 + 15;                    // causal horizon

  stage(0, 0);
  int cur = 0;
  for (int kt0 = 0; kt0 <= kt_end; kt0 += 64) {
    if (kt0 + 64 <= kt_end) {
      stage(cur ^ 1, kt0 + 64);                  // prefetch next block
      async_wait_one_block();                    // current block resident
    } else {
      async_wait_all();
    }
    const f16_t* Kt = &Kbuf[cur][0];
    const f16_t* Vl = &Vbuf[cur][0];

    // ---- S = Q @ K^T over this 16 x 64 key block (4 tiles, Dh in 2 chunks)
    v8f S[4];
#pragma unroll
    for (int jn = 0; jn < 4; ++jn) {
      v8f acc = zero;
#pragma unroll
      for (int c = 0; c < 2; ++c) {
        v16h bf = make_frag(Kt + (jn * 16 + r) * 64 + c * 32 + hf * 8);
        acc = wmma_f16(qf[c], bf, acc);
      }
      S[jn] = acc;
    }

    // ---- scale + causal mask + per-row block max
    float bmax[8];
#pragma unroll
    for (int i = 0; i < 8; ++i) bmax[i] = -3.0e38f;
#pragma unroll
    for (int jn = 0; jn < 4; ++jn) {
      const int kg = kt0 + jn * 16 + r;          // key index (col N = r)
#pragma unroll
      for (int i = 0; i < 8; ++i) {
        const int qg = q0 + hf * 8 + i;          // query row
        float s = S[jn][i] * scale;
        s = (kg > qg) ? -3.0e38f : s;
        S[jn][i] = s;
        bmax[i] = fmaxf(bmax[i], s);
      }
    }
#pragma unroll
    for (int msk = 1; msk <= 8; msk <<= 1)
#pragma unroll
      for (int i = 0; i < 8; ++i)
        bmax[i] = fmaxf(bmax[i], __shfl_xor(bmax[i], msk, 32));

    // ---- online softmax state update
    float alpha[8], rsum[8];
#pragma unroll
    for (int i = 0; i < 8; ++i) {
      const float mn = fmaxf(mrow[i], bmax[i]);
      alpha[i] = __expf(mrow[i] - mn);
      mrow[i] = mn;
      rsum[i] = 0.0f;
    }
#pragma unroll
    for (int jn = 0; jn < 4; ++jn) {
#pragma unroll
      for (int i = 0; i < 8; ++i) {
        const float p = __expf(S[jn][i] - mrow[i]);
        rsum[i] += p;
        Pbuf[hf * 8 + i][jn * 16 + r] = (f16_t)p;   // C-layout -> LDS
      }
    }
#pragma unroll
    for (int msk = 1; msk <= 8; msk <<= 1)
#pragma unroll
      for (int i = 0; i < 8; ++i)
        rsum[i] += __shfl_xor(rsum[i], msk, 32);
#pragma unroll
    for (int i = 0; i < 8; ++i) lrow[i] = lrow[i] * alpha[i] + rsum[i];
#pragma unroll
    for (int jd = 0; jd < 4; ++jd)
#pragma unroll
      for (int i = 0; i < 8; ++i) O[jd][i] *= alpha[i];

    __syncthreads();  // single-wave WG: ordering hint only

    // ---- O += P @ V : A = P (A-layout rows from LDS), B = V^T rows (d),
    //      contraction over keys in 2 chunks of 32.
#pragma unroll
    for (int jd = 0; jd < 4; ++jd) {
#pragma unroll
      for (int c = 0; c < 2; ++c) {
        v16h af = make_frag(&Pbuf[r][c * 32 + hf * 8]);
        v16h bf = make_frag(Vl + (jd * 16 + r) * 64 + c * 32 + hf * 8);
        O[jd] = wmma_f16(af, bf, O[jd]);
      }
    }
    __syncthreads();
    cur ^= 1;
  }

  // ---- normalize and store merged-head layout [B, T, C] (f16)
#pragma unroll
  for (int jd = 0; jd < 4; ++jd) {
#pragma unroll
    for (int i = 0; i < 8; ++i) {
      const int t = q0 + hf * 8 + i;
      const float v = O[jd][i] / lrow[i];
      attn[((size_t)(b * SEQ + t)) * D_MODEL + h * 64 + jd * 16 + r] = (f16_t)v;
    }
  }
}

// ---------------------------------------------------------------------------
// Output projection: [8192,1024](f16) x [1024,1024](f16,T) + b_out -> fp32 out
// Same ping-pong software-pipelined 64x64-per-wave structure as the QKV GEMM.
// ---------------------------------------------------------------------------
__global__ __launch_bounds__(32) void out_gemm_kernel(
    const f16_t* __restrict__ ah, const f16_t* __restrict__ wt,
    const float* __restrict__ bias, float* __restrict__ out) {
  const int lane = threadIdx.x & 31;
  const int r = lane & 15;
  const int hf = lane >> 4;
  const int m0 = blockIdx.x * 64;
  const int n0 = blockIdx.y * 64;

  v8f zero = {};
  v8f acc[4][4];
#pragma unroll
  for (int i = 0; i < 4; ++i)
#pragma unroll
    for (int j = 0; j < 4; ++j) acc[i][j] = zero;

  v16h a0[4], b0[4], a1[4], b1[4];

  auto load_set = [&](v16h* A, v16h* B, int kk) {
#pragma unroll
    for (int i = 0; i < 4; ++i)
      A[i] = load_frag(ah, D_MODEL, m0 + i * 16 + r, kk, hf);
#pragma unroll
    for (int j = 0; j < 4; ++j)
      B[j] = load_frag(wt, D_MODEL, n0 + j * 16 + r, kk, hf);
  };
  auto mma_set = [&](const v16h* A, const v16h* B) {
#pragma unroll
    for (int i = 0; i < 4; ++i)
#pragma unroll
      for (int j = 0; j < 4; ++j)
        acc[i][j] = wmma_f16(A[i], B[j], acc[i][j]);
  };

  load_set(a0, b0, 0);
  for (int k0 = 0; k0 < D_MODEL; k0 += 64) {
    load_set(a1, b1, k0 + 32);
    __builtin_prefetch(ah + (size_t)(m0 + r) * D_MODEL + k0 + 64, 0, 1);
    __builtin_prefetch(wt + (size_t)(n0 + r) * D_MODEL + k0 + 64, 0, 1);
    mma_set(a0, b0);
    if (k0 + 64 < D_MODEL) load_set(a0, b0, k0 + 64);
    mma_set(a1, b1);
  }

#pragma unroll
  for (int j = 0; j < 4; ++j) {
    const int n = n0 + j * 16 + r;
    const float bv = bias[n];
#pragma unroll
    for (int i = 0; i < 4; ++i) {
#pragma unroll
      for (int e = 0; e < 8; ++e) {
        const int m = m0 + i * 16 + hf * 8 + e;
        out[(size_t)m * D_MODEL + n] = acc[i][j][e] + bv;
      }
    }
  }
}

// ---------------------------------------------------------------------------
// Host-side launcher
// ---------------------------------------------------------------------------
extern "C" void kernel_launch(void* const* d_in, const int* in_sizes, int n_in,
                              void* d_out, int out_size, void* d_ws,
                              size_t ws_size, hipStream_t stream) {
  (void)in_sizes; (void)n_in; (void)out_size; (void)ws_size;

  const float* x     = (const float*)d_in[0];
  const float* W_qkv = (const float*)d_in[1];
  const float* b_qkv = (const float*)d_in[2];
  const float* W_out = (const float*)d_in[3];
  const float* b_out = (const float*)d_in[4];
  float* out = (float*)d_out;

  char* ws = (char*)d_ws;
  size_t off = 0;
  auto carve = [&](size_t bytes) -> char* {
    char* p = ws + off;
    off += (bytes + 255) & ~(size_t)255;
    return p;
  };
  const size_t bhtd = (size_t)BATCH * N_HEADS * SEQ * D_HEAD;  // 8.39M elems
  f16_t* xh    = (f16_t*)carve((size_t)M_TOTAL * D_MODEL * 2);
  f16_t* wqkvT = (f16_t*)carve((size_t)3 * D_MODEL * D_MODEL * 2);
  f16_t* woutT = (f16_t*)carve((size_t)D_MODEL * D_MODEL * 2);
  f16_t* qB    = (f16_t*)carve(bhtd * 2);
  f16_t* kB    = (f16_t*)carve(bhtd * 2);
  f16_t* vtB   = (f16_t*)carve(bhtd * 2);
  f16_t* attnB = (f16_t*)carve((size_t)M_TOTAL * D_MODEL * 2);

  // 1) fp32 -> f16 (x) ; fp32 -> f16 + transpose (weights, so WMMA B-frags
  //    read contiguous K)
  {
    int n = M_TOTAL * D_MODEL;
    cvt_f32_to_f16<<<(n + 255) / 256, 256, 0, stream>>>(x, xh, n);
    int nw1 = D_MODEL * 3 * D_MODEL;
    cvt_transpose_f32_to_f16<<<(nw1 + 255) / 256, 256, 0, stream>>>(
        W_qkv, wqkvT, D_MODEL, 3 * D_MODEL);
    int nw2 = D_MODEL * D_MODEL;
    cvt_transpose_f32_to_f16<<<(nw2 + 255) / 256, 256, 0, stream>>>(
        W_out, woutT, D_MODEL, D_MODEL);
  }

  // 2) QKV projection GEMM (writes per-head Q, K and transposed V)
  qkv_gemm_kernel<<<dim3(M_TOTAL / 64, (3 * D_MODEL) / 64), 32, 0, stream>>>(
      xh, wqkvT, b_qkv, qB, kB, vtB);

  // 3) causal flash attention with async-LDS double buffering
  attn_kernel<<<BATCH * N_HEADS * (SEQ / 16), 32, 0, stream>>>(qB, kB, vtB,
                                                               attnB);

  // 4) output projection GEMM -> fp32 result
  out_gemm_kernel<<<dim3(M_TOTAL / 64, D_MODEL / 64), 32, 0, stream>>>(
      attnB, woutT, b_out, out);
}
